// VectorQuantizer_5961414606896
// MI455X (gfx1250) — matrix operand order repaired and checked by reading, not compiled, not run on hardware
//
#include <hip/hip_runtime.h>

// ---------------------------------------------------------------------------
// VQ-VAE vector quantizer for MI455X (gfx1250), wave32 + WMMA bf16.
//   x: [131072, 64] fp32, E: [64, 512] fp32.
//   scores = ||e_k||^2 - 2 * x.E  (||x||^2 constant per row -> dropped)
//   codes = argmin_k, out = E[:, code], loss = 1.25*mean((x-q)^2)
// ---------------------------------------------------------------------------

typedef __bf16 bf16_t;
typedef bf16_t v16bf __attribute__((ext_vector_type(16)));
typedef float  v8f   __attribute__((ext_vector_type(8)));

#define D_DIM   64
#define K_DIM   512
#define NTILES  32          // 512 / 16 column tiles
#define WAVES   8
#define SUBT    4           // 16-row WMMA subtiles per wave
#define ROWS_PER_WAVE (16 * SUBT)    // 64 rows: 4x B-fragment reuse
#define N_ROWS  131072
#define BPACK_ELEMS (NTILES * 2 * 32 * 16)   // 32768 bf16 = 64 KB

// ---------------------------------------------------------------------------
// Prep: pack E into bf16 WMMA B-fragment layout, compute ||e_k||^2, zero loss.
// Bpack linear index = ((t*2 + s)*32 + lane)*16 + j
//   lane n   (0..15): element (K = 32*s + j,      N = 16*t + n)
//   lane n+16       : element (K = 32*s + 16 + j, N = 16*t + n)
// ---------------------------------------------------------------------------
__global__ __launch_bounds__(256)
void vq_prep(const float* __restrict__ E, bf16_t* __restrict__ Bpack,
             float* __restrict__ enorm, float* __restrict__ lossSlot) {
    int tid = blockIdx.x * blockDim.x + threadIdx.x;
    if (tid < BPACK_ELEMS) {
        int j    = tid & 15;
        int lane = (tid >> 4) & 31;
        int s    = (tid >> 9) & 1;
        int t    = tid >> 10;
        int d    = 32 * s + 16 * (lane >> 4) + j;   // K index (0..63)
        int c    = 16 * t + (lane & 15);            // N index (0..511)
        Bpack[tid] = (bf16_t)E[d * K_DIM + c];
    }
    if (tid < K_DIM) {
        float sum = 0.0f;
        #pragma unroll
        for (int d = 0; d < D_DIM; ++d) {
            float v = E[d * K_DIM + tid];
            sum += v * v;
        }
        enorm[tid] = sum;
    }
    if (tid == 0) *lossSlot = 0.0f;
}

// ---------------------------------------------------------------------------
// Main kernel: 256 threads = 8 waves; each wave handles 64 rows.
// ---------------------------------------------------------------------------
__global__ __launch_bounds__(256)
void vq_main(const float* __restrict__ x, const float* __restrict__ E,
             const bf16_t* __restrict__ Bpack, const float* __restrict__ enorm,
             float* __restrict__ out, float* __restrict__ lossSlot) {
    __shared__ uint4 sBraw[BPACK_ELEMS * 2 / 16];   // 64 KB bf16 B pack
    __shared__ float sEn[K_DIM];                    // 2 KB
    __shared__ int   sCodes[WAVES][ROWS_PER_WAVE];  // 2 KB

    const int tid  = threadIdx.x;
    const int lane = tid & 31;
    const int wave = tid >> 5;

    // ---- Cooperative stage of B pack + norms into LDS -------------------
    {
        const uint4* gB = (const uint4*)Bpack;
        #pragma unroll
        for (int i = 0; i < (BPACK_ELEMS * 2 / 16) / 256; ++i)
            sBraw[tid + i * 256] = gB[tid + i * 256];
        #pragma unroll
        for (int i = 0; i < K_DIM / 256; ++i)
            sEn[tid + i * 256] = enorm[tid + i * 256];
    }
    __syncthreads();
    const bf16_t* sB = (const bf16_t*)sBraw;

    const int waveGlobal = blockIdx.x * WAVES + wave;      // 0..2047
    const int rowBase    = waveGlobal * ROWS_PER_WAVE;     // 0..131008

    const int m  = lane & 15;
    const int hi = lane >> 4;

    // ---- Build A fragments (contiguous k order, matches Bpack), a[u][s] ---------------
    // lane m, hi: av[0..7] = K = 32s + 16*hi + 0..7
    //             av[8..15]= K = 32s + 16*hi + 8 + 0..7
    v16bf a[SUBT][2];
    #pragma unroll
    for (int u = 0; u < SUBT; ++u) {
        const float* xr = x + (rowBase + 16 * u + m) * D_DIM;
        #pragma unroll
        for (int s = 0; s < 2; ++s) {
            const float* p0 = xr + 32 * s + 16 * hi;
            const float* p1 = xr + 32 * s + 16 * hi + 8;
            float f[16];
            *(float4*)(f + 0)  = *(const float4*)(p0);
            *(float4*)(f + 4)  = *(const float4*)(p0 + 4);
            *(float4*)(f + 8)  = *(const float4*)(p1);
            *(float4*)(f + 12) = *(const float4*)(p1 + 4);
            v16bf av;
            #pragma unroll
            for (int j = 0; j < 16; ++j) av[j] = (bf16_t)f[j];
            a[u][s] = av;
        }
    }

    // ---- Score all 512 codes via WMMA, keep running argmin --------------
    float best[SUBT][8];
    int   bidx[SUBT][8];
    #pragma unroll
    for (int u = 0; u < SUBT; ++u)
        #pragma unroll
        for (int r = 0; r < 8; ++r) { best[u][r] = 3.4e38f; bidx[u][r] = 0; }

    for (int t = 0; t < NTILES; ++t) {
        v16bf b0 = *(const v16bf*)(sB + ((2 * t + 0) * 32 + lane) * 16);
        v16bf b1 = *(const v16bf*)(sB + ((2 * t + 1) * 32 + lane) * 16);
        const int   colIdx = 16 * t + (lane & 15);
        const float en     = sEn[colIdx];

        #pragma unroll
        for (int u = 0; u < SUBT; ++u) {
            v8f c = {};
            c = __builtin_amdgcn_wmma_f32_16x16x32_bf16(
                    false, a[u][0], false, b0, (short)0, c, false, false);
            c = __builtin_amdgcn_wmma_f32_16x16x32_bf16(
                    false, a[u][1], false, b1, (short)0, c, false, false);
            #pragma unroll
            for (int r = 0; r < 8; ++r) {
                float v  = en - 2.0f * c[r];
                bool  lt = v < best[u][r];                 // strict < keeps lowest index
                best[u][r] = lt ? v : best[u][r];
                bidx[u][r] = lt ? colIdx : bidx[u][r];
            }
        }
    }

    // ---- Cross-lane argmin within each 16-lane half (C-matrix striping) --
    // C layout: VGPR r, lanes 0-15 -> M=r ; lanes 16-31 -> M=8+r ; N=lane%16.
    #pragma unroll
    for (int u = 0; u < SUBT; ++u) {
        #pragma unroll
        for (int r = 0; r < 8; ++r) {
            float v   = best[u][r];
            int   idx = bidx[u][r];
            #pragma unroll
            for (int off = 1; off < 16; off <<= 1) {
                float ov = __shfl_xor(v, off, 16);
                int   oi = __shfl_xor(idx, off, 16);
                if (ov < v || (ov == v && oi < idx)) { v = ov; idx = oi; }
            }
            if ((lane & 15) == 0)
                sCodes[wave][16 * u + 8 * hi + r] = idx;
        }
    }
    __syncthreads();

    // ---- Output: gather fp32 codebook columns, write out, loss ----------
    // Coalesced: iter i covers dwords [i*128, i*128+128) of the 64x64 tile.
    float lsum = 0.0f;
    const int base = rowBase * D_DIM;
    #pragma unroll 4
    for (int i = 0; i < ROWS_PER_WAVE * D_DIM / 128; ++i) {
        int off  = i * 128 + lane * 4;     // dword offset in tile
        int row  = off >> 6;               // 0..63
        int col  = off & 63;               // multiple of 4
        int code = sCodes[wave][row];
        float4 xv = *(const float4*)(x + base + off);
        float4 q;
        q.x = E[(col + 0) * K_DIM + code];
        q.y = E[(col + 1) * K_DIM + code];
        q.z = E[(col + 2) * K_DIM + code];
        q.w = E[(col + 3) * K_DIM + code];
        *(float4*)(out + base + off) = q;
        float dx = xv.x - q.x, dy = xv.y - q.y, dz = xv.z - q.z, dw = xv.w - q.w;
        lsum += dx * dx + dy * dy + dz * dz + dw * dw;
    }
    // full-wave reduction (wave32)
    #pragma unroll
    for (int off = 1; off < 32; off <<= 1)
        lsum += __shfl_xor(lsum, off, 32);
    if (lane == 0)
        atomicAdd(lossSlot, lsum * (1.25f / 8388608.0f));   // 1.25 * mean
}

// ---------------------------------------------------------------------------
extern "C" void kernel_launch(void* const* d_in, const int* in_sizes, int n_in,
                              void* d_out, int out_size, void* d_ws, size_t ws_size,
                              hipStream_t stream) {
    const float* x = (const float*)d_in[0];   // 32*64*64*64 fp32
    const float* E = (const float*)d_in[1];   // 64*512 fp32
    float* out = (float*)d_out;               // 8388608 out + 1 loss
    (void)in_sizes; (void)n_in; (void)ws_size;

    bf16_t* Bpack    = (bf16_t*)d_ws;                       // 64 KB
    float*  enorm    = (float*)((char*)d_ws + BPACK_ELEMS * 2); // 2 KB
    float*  lossSlot = out + (out_size - 1);

    vq_prep<<<BPACK_ELEMS / 256, 256, 0, stream>>>(E, Bpack, enorm, lossSlot);

    // 131072 rows / 64 rows-per-wave / 8 waves-per-block = 256 blocks
    vq_main<<<(N_ROWS / ROWS_PER_WAVE) / WAVES, 256, 0, stream>>>(
        x, E, Bpack, enorm, out, lossSlot);
}
